// StyleDecorator_8778913153275
// MI455X (gfx1250) — matrix operand order repaired and checked by reading, not compiled, not run on hardware
//
#include <hip/hip_runtime.h>
#include <hip/hip_bf16.h>

// ---------------------------------------------------------------------------
// Style-decorator pipeline for MI455X (gfx1250, wave32, WMMA + TDM).
//   1. mean-center features -> bf16
//   2. cov = f f^T / (N-1)                        (WMMA, direct-global frags)
//   3. coupled Newton-Schulz -> cov^{+/-1/2}      (3 WMMA matmuls / iter)
//   4. whiten apply -> padded bf16 feature maps   (WMMA, B staged by TDM)
//   5. patch-norm reciprocal
//   6. fused score GEMM + running argmax          (WMMA, LDS-table staging)
//   7. argmax merge, overlap-add reassembly
//   8. color apply + style mean -> d_out          (WMMA, B staged by TDM)
// Workspace requirement: ~70 MB.
// ---------------------------------------------------------------------------

#define B_   2
#define C_   512
#define H_   64
#define W_   64
#define N_   4096      // H*W
#define M_   4608      // C*9
#define HP_  66        // H + 2*PAD
#define PLANE_ (HP_ * HP_)
#define NS_ITERS 12
#define PSPLIT 4
#define QBLK   64
#define PPANEL 128
#define MSTEP  32

typedef __attribute__((ext_vector_type(16))) __bf16 v16bf;
typedef __attribute__((ext_vector_type(8)))  float  v8f;
typedef __attribute__((ext_vector_type(4)))  unsigned int u32x4;
typedef __attribute__((ext_vector_type(4)))  int i32x4;
typedef __attribute__((ext_vector_type(8)))  int i32x8;

struct H16 { unsigned short h[16]; };

__device__ __forceinline__ unsigned short f2bf(float f) {
  unsigned u = __builtin_bit_cast(unsigned, f);
  u += 0x7FFFu + ((u >> 16) & 1u);             // round-to-nearest-even
  return (unsigned short)(u >> 16);
}
__device__ __forceinline__ float bf2f(unsigned short h) {
  unsigned u = ((unsigned)h) << 16;
  return __builtin_bit_cast(float, u);
}

__device__ __forceinline__ v8f wmma_bf16(v16bf a, v16bf b, v8f c) {
  return __builtin_amdgcn_wmma_f32_16x16x32_bf16(
      false, a, false, b, (short)0, c, false, false);
}

// ---- Tensor Data Mover: 2D u16 tile (rows x cols) from a row-major matrix
// (row stride `stride_elems`, `tensor_rows` total rows for OOB) into LDS.
// D# layout per CDNA5 ISA ch.8 (group0: count/lds/global/type, group1:
// data_size, tensor_dim0/1, tile_dim0/1, tensor_dim0_stride).
__device__ __forceinline__ void tdm_load_2d_u16(unsigned lds_off,
                                                const void* gptr,
                                                unsigned rows, unsigned cols,
                                                unsigned stride_elems,
                                                unsigned tensor_rows) {
  unsigned long long ga = (unsigned long long)(size_t)gptr;
  u32x4 g0;
  g0[0] = 1u;                                        // count=1, user descriptor
  g0[1] = lds_off;                                   // lds_addr (bytes)
  g0[2] = (unsigned)(ga & 0xffffffffu);              // global_addr[31:0]
  g0[3] = (unsigned)((ga >> 32) & 0x1ffffffu)        // global_addr[56:32]
          | (2u << 30);                              // type=2 ("image")
  i32x8 g1 = {0, 0, 0, 0, 0, 0, 0, 0};
  g1[0] = (int)(1u << 16);                           // data_size=1 -> 2 bytes
  g1[1] = (int)((stride_elems & 0xffffu) << 16);     // tensor_dim0[15:0]
  g1[2] = (int)(((stride_elems >> 16) & 0xffffu)     // tensor_dim0[31:16]
                | ((tensor_rows & 0xffffu) << 16));  // tensor_dim1[15:0]
  g1[3] = (int)(((tensor_rows >> 16) & 0xffffu)      // tensor_dim1[31:16]
                | (cols << 16));                     // tile_dim0
  g1[4] = (int)rows;                                 // tile_dim1 (tile_dim2=0)
  g1[5] = (int)stride_elems;                         // tensor_dim0_stride[31:0]
  g1[6] = 0;                                         // stride hi / dim1_stride
  i32x4 gz = {0, 0, 0, 0};
#if defined(__clang_major__) && __clang_major__ >= 23
  i32x8 gz8 = {0, 0, 0, 0, 0, 0, 0, 0};
  __builtin_amdgcn_tensor_load_to_lds(g0, g1, gz, gz, gz8, 0);
#else
  __builtin_amdgcn_tensor_load_to_lds(g0, g1, gz, gz, 0);
#endif
}

// A fragment: 16x32 (MxK) tile, row-major u16(bf16) storage, leading dim ld.
__device__ __forceinline__ v16bf frag_a_u16(const unsigned short* base, int ld) {
  int lane = threadIdx.x & 31;
  int row  = lane & 15;
  int kb   = (lane & 16) ? 16 : 0;
  const unsigned short* r = base + row * ld;
  H16 t;
#pragma unroll
  for (int i = 0; i < 16; ++i) t.h[i] = r[kb + i];
  return __builtin_bit_cast(v16bf, t);
}
__device__ __forceinline__ v16bf frag_a_f32(const float* base, int ld) {
  int lane = threadIdx.x & 31;
  int row  = lane & 15;
  int kb   = (lane & 16) ? 16 : 0;
  const float* r = base + row * ld;
  H16 t;
#pragma unroll
  for (int i = 0; i < 16; ++i) t.h[i] = f2bf(r[kb + i]);
  return __builtin_bit_cast(v16bf, t);
}

// B fragment: 32x16 (KxN) tile from "NT" storage (row n holds K-contiguous).
__device__ __forceinline__ v16bf frag_b_nt_u16(const unsigned short* base, int ld) {
  int lane = threadIdx.x & 31;
  int n    = lane & 15;
  int kb   = (lane & 16) ? 16 : 0;
  const unsigned short* r = base + n * ld + kb;
  H16 t;
#pragma unroll
  for (int i = 0; i < 16; ++i) t.h[i] = r[i];
  return __builtin_bit_cast(v16bf, t);
}
// B fragment from K-major (row-major [k][n]) storage, leading dim ld.
__device__ __forceinline__ v16bf frag_b_nn_u16(const unsigned short* base, int ld) {
  int lane = threadIdx.x & 31;
  int n    = lane & 15;
  int kb   = (lane & 16) ? 16 : 0;
  H16 t;
#pragma unroll
  for (int i = 0; i < 16; ++i) t.h[i] = base[(size_t)(kb + i) * ld + n];
  return __builtin_bit_cast(v16bf, t);
}
__device__ __forceinline__ v16bf frag_b_nn_f32(const float* base, int ld) {
  int lane = threadIdx.x & 31;
  int n    = lane & 15;
  int kb   = (lane & 16) ? 16 : 0;
  H16 t;
#pragma unroll
  for (int i = 0; i < 16; ++i) t.h[i] = f2bf(base[(size_t)(kb + i) * ld + n]);
  return __builtin_bit_cast(v16bf, t);
}

// ---------------------------------------------------------------------------
__global__ void fill_u16_kernel(unsigned short* p, size_t n) {
  size_t i = (size_t)blockIdx.x * blockDim.x + threadIdx.x;
  if (i < n) p[i] = 0;
}

__global__ __launch_bounds__(256)
void mean_center_kernel(const float* __restrict__ content,
                        const float* __restrict__ style,
                        unsigned short* fc_bf, unsigned short* fs_bf,
                        float* meanC, float* meanS) {
  int bc = blockIdx.x;
  const float* src = (blockIdx.y ? style : content) + (size_t)bc * N_;
  unsigned short* dst = (blockIdx.y ? fs_bf : fc_bf) + (size_t)bc * N_;
  float* mo = blockIdx.y ? meanS : meanC;
  __shared__ float red[256];
  float s = 0.f;
  for (int i = threadIdx.x; i < N_; i += 256) s += src[i];
  red[threadIdx.x] = s; __syncthreads();
  for (int st = 128; st > 0; st >>= 1) {
    if (threadIdx.x < st) red[threadIdx.x] += red[threadIdx.x + st];
    __syncthreads();
  }
  float mean = red[0] * (1.0f / N_);
  if (threadIdx.x == 0) mo[bc] = mean;
  for (int i = threadIdx.x; i < N_; i += 256) dst[i] = f2bf(src[i] - mean);
}

// cov[mat] = f f^T / (N-1); mat = b*2 + tensor.  One wave per 16x16 tile.
__global__ __launch_bounds__(256)
void cov_kernel(const unsigned short* fc_bf, const unsigned short* fs_bf,
                float* cov) {
  int mat = blockIdx.z;
  const unsigned short* f =
      ((mat & 1) ? fs_bf : fc_bf) + (size_t)(mat >> 1) * C_ * N_;
  int wave = threadIdx.x >> 5;
  int tile = blockIdx.x * 8 + wave;
  int tm = tile >> 5, tn = tile & 31;
  v8f acc = {};
  for (int k0 = 0; k0 < N_; k0 += MSTEP) {
    v16bf a = frag_a_u16(f + (size_t)(tm * 16) * N_ + k0, N_);
    v16bf b = frag_b_nt_u16(f + (size_t)(tn * 16) * N_ + k0, N_);
    acc = wmma_bf16(a, b, acc);
  }
  int lane = threadIdx.x & 31;
  int col  = tn * 16 + (lane & 15);
  int rb   = (lane & 16) ? 8 : 0;
  float scale = 1.0f / (float)(N_ - 1);
  float* out = cov + (size_t)mat * C_ * C_;
#pragma unroll
  for (int r = 0; r < 8; ++r)
    out[(size_t)(tm * 16 + rb + r) * C_ + col] = acc[r] * scale;
}

__global__ void trace_kernel(const float* cov, float* tr) {
  int mat = blockIdx.x;
  __shared__ float red[256];
  float s = 0.f;
  for (int i = threadIdx.x; i < C_; i += 256)
    s += cov[(size_t)mat * C_ * C_ + (size_t)i * C_ + i];
  red[threadIdx.x] = s; __syncthreads();
  for (int st = 128; st > 0; st >>= 1) {
    if (threadIdx.x < st) red[threadIdx.x] += red[threadIdx.x + st];
    __syncthreads();
  }
  if (threadIdx.x == 0) tr[mat] = red[0];
}

__global__ void init_yz_kernel(const float* cov, const float* tr,
                               float* Y, float* Z) {
  size_t i = (size_t)blockIdx.x * 256 + threadIdx.x;
  if (i >= (size_t)4 * C_ * C_) return;
  int mat = (int)(i / (C_ * C_));
  size_t rc = i % (C_ * C_);
  Y[i] = cov[i] / tr[mat];
  Z[i] = ((rc / C_) == (rc % C_)) ? 1.0f : 0.0f;
}

// D = alpha * (A @ B) + beta_diag * I, for 4 batched 512x512 f32 matrices.
__global__ __launch_bounds__(256)
void ns_matmul_kernel(const float* A, const float* Bm, float* D,
                      float alpha, float beta_diag) {
  int mat = blockIdx.z;
  const float* Ab = A  + (size_t)mat * C_ * C_;
  const float* Bb = Bm + (size_t)mat * C_ * C_;
  float*       Db = D  + (size_t)mat * C_ * C_;
  int wave = threadIdx.x >> 5;
  int tile = blockIdx.x * 8 + wave;
  int tm = tile >> 5, tn = tile & 31;
  v8f acc = {};
  for (int k0 = 0; k0 < C_; k0 += MSTEP) {
    v16bf a = frag_a_f32(Ab + (size_t)(tm * 16) * C_ + k0, C_);
    v16bf b = frag_b_nn_f32(Bb + (size_t)k0 * C_ + tn * 16, C_);
    acc = wmma_bf16(a, b, acc);
  }
  int lane = threadIdx.x & 31;
  int col  = tn * 16 + (lane & 15);
  int rb   = (lane & 16) ? 8 : 0;
#pragma unroll
  for (int r = 0; r < 8; ++r) {
    int row = tm * 16 + rb + r;
    Db[(size_t)row * C_ + col] = alpha * acc[r] + ((row == col) ? beta_diag : 0.f);
  }
}

// padded(c, y+1, x+1) = (cov^{-1/2} @ f_centered) in bf16.
// B panel (32 k-rows x 128 cols) staged per k-step via TDM -> LDS.
__global__ __launch_bounds__(256)
void whiten_apply_kernel(const float* Z, const float* tr,
                         const unsigned short* fc_bf, const unsigned short* fs_bf,
                         unsigned short* cpad, unsigned short* spad) {
  int mat = blockIdx.z;                         // b*2 + tensor
  int b = mat >> 1, tensor = mat & 1;
  const float* Zb = Z + (size_t)mat * C_ * C_;
  const unsigned short* f = (tensor ? fs_bf : fc_bf) + (size_t)b * C_ * N_;
  unsigned short* outp = (tensor ? spad : cpad) + (size_t)b * C_ * PLANE_;
  float scale = rsqrtf(tr[mat]);
  int wave = threadIdx.x >> 5;
  int tile = blockIdx.x * 8 + wave;             // 32*256 tiles; block: 8 consec tn
  int tm = tile >> 8, tn = tile & 255;
  int tn0 = (blockIdx.x * 8) & 255;

  __shared__ unsigned short ldsB[MSTEP * 128];  // 8 KB K-major panel

  v8f acc = {};
  for (int k0 = 0; k0 < C_; k0 += MSTEP) {
    if (threadIdx.x < 32) {
      tdm_load_2d_u16((unsigned)(size_t)&ldsB[0],
                      f + (size_t)k0 * N_ + (size_t)tn0 * 16,
                      /*rows=*/MSTEP, /*cols=*/128,
                      /*stride=*/N_, /*tensor_rows=*/C_);
      __builtin_amdgcn_s_wait_tensorcnt(0);
    }
    __syncthreads();
    v16bf a = frag_a_f32(Zb + (size_t)(tm * 16) * C_ + k0, C_);
    v16bf bfr = frag_b_nn_u16(ldsB + (tn - tn0) * 16, 128);
    acc = wmma_bf16(a, bfr, acc);
    __syncthreads();
  }
  int lane = threadIdx.x & 31;
  int col  = tn * 16 + (lane & 15);
  int y = col >> 6, x = col & 63;
  int rb = (lane & 16) ? 8 : 0;
#pragma unroll
  for (int r = 0; r < 8; ++r) {
    int c = tm * 16 + rb + r;
    outp[((size_t)c * HP_ + (y + 1)) * HP_ + (x + 1)] = f2bf(acc[r] * scale);
  }
}

// rknorm[b,p] = 1 / (||style patch p||_2 + 1e-5)
__global__ __launch_bounds__(128)
void knorm_kernel(const unsigned short* spad, float* rknorm) {
  int b = blockIdx.x / N_;
  int p = blockIdx.x % N_;
  int pbase = (p >> 6) * HP_ + (p & 63);
  const unsigned short* sp = spad + (size_t)b * C_ * PLANE_;
  __shared__ float red[128];
  float s = 0.f;
  for (int m = threadIdx.x; m < M_; m += 128) {
    int c = m / 9, t9 = m - c * 9;
    int oy = t9 / 3, ox = t9 - oy * 3;
    float v = bf2f(sp[(size_t)(pbase + c * PLANE_ + oy * HP_ + ox)]);
    s += v * v;
  }
  red[threadIdx.x] = s; __syncthreads();
  for (int st = 64; st > 0; st >>= 1) {
    if (threadIdx.x < st) red[threadIdx.x] += red[threadIdx.x + st];
    __syncthreads();
  }
  if (threadIdx.x == 0) rknorm[blockIdx.x] = 1.0f / (sqrtf(red[0]) + 1e-5f);
}

__device__ __forceinline__ void upd_argmax(const v8f& acc, const float* rkv,
                                           int prow0, float& best, int& bidx) {
#pragma unroll
  for (int r = 0; r < 8; ++r) {
    float s = acc[r] * rkv[r];
    if (s > best) { best = s; bidx = prow0 + r; }
  }
}

// Fused score GEMM + running argmax.  Gather addresses decompose as
// ppos[p] + moff[m]; moff (4608 ints) lives in LDS so the 144x-iterated
// staging loop is pure ds_load + global_load_u16 + ds_store.
__global__ __launch_bounds__(256)
void score_argmax_kernel(const unsigned short* cpad, const unsigned short* spad,
                         const float* rknorm, float* bestOut, int* idxOut) {
  int b      = blockIdx.z;
  int qbase  = blockIdx.x * QBLK;
  int pStart = blockIdx.y * (N_ / PSPLIT);
  const unsigned short* cp = cpad + (size_t)b * C_ * PLANE_;
  const unsigned short* sp = spad + (size_t)b * C_ * PLANE_;
  const float* rk = rknorm + (size_t)b * N_;

  __shared__ int smoff[M_];                        // 18 KB channel-plane offsets
  __shared__ int qpos[QBLK];
  __shared__ int ppos[PPANEL];
  __shared__ unsigned short ldsK[PPANEL * MSTEP];  // 8 KB style panel
  __shared__ unsigned short ldsC[QBLK * MSTEP];    // 4 KB content panel
  __shared__ float redB[8 * QBLK];
  __shared__ int   redI[8 * QBLK];

  int tid = threadIdx.x, wave = tid >> 5, lane = tid & 31;

  for (int m = tid; m < M_; m += 256) {            // one-time table build
    int c = m / 9, t9 = m - c * 9;
    int oy = t9 / 3, ox = t9 - oy * 3;
    smoff[m] = c * PLANE_ + oy * HP_ + ox;
  }
  if (tid < QBLK) {
    int q = qbase + tid;
    qpos[tid] = (q >> 6) * HP_ + (q & 63);
  }

  float best[4] = {-1e30f, -1e30f, -1e30f, -1e30f};
  int   bidx[4] = {0, 0, 0, 0};

  for (int pp0 = 0; pp0 < N_ / PSPLIT; pp0 += PPANEL) {
    int pbase = pStart + pp0;
    if (tid < PPANEL) {
      int p = pbase + tid;
      ppos[tid] = (p >> 6) * HP_ + (p & 63);
    }
    v8f acc0 = {}, acc1 = {}, acc2 = {}, acc3 = {};
    for (int m0 = 0; m0 < M_; m0 += MSTEP) {
      __syncthreads();
#pragma unroll
      for (int e = tid; e < PPANEL * MSTEP; e += 256)
        ldsK[e] = sp[(size_t)(ppos[e >> 5] + smoff[m0 + (e & 31)])];
#pragma unroll
      for (int e = tid; e < QBLK * MSTEP; e += 256)
        ldsC[e] = cp[(size_t)(qpos[e >> 5] + smoff[m0 + (e & 31)])];
      if (m0 + MSTEP < M_)   // gfx1250 global_prefetch_b8 on next channel block
        __builtin_prefetch(sp + (size_t)((m0 + MSTEP) / 9) * PLANE_, 0, 1);
      __syncthreads();
      v16bf a = frag_a_u16(ldsK + (size_t)(16 * wave) * MSTEP, MSTEP);
      acc0 = wmma_bf16(a, frag_b_nt_u16(ldsC +  0 * MSTEP, MSTEP), acc0);
      acc1 = wmma_bf16(a, frag_b_nt_u16(ldsC + 16 * MSTEP, MSTEP), acc1);
      acc2 = wmma_bf16(a, frag_b_nt_u16(ldsC + 32 * MSTEP, MSTEP), acc2);
      acc3 = wmma_bf16(a, frag_b_nt_u16(ldsC + 48 * MSTEP, MSTEP), acc3);
    }
    int rb = (lane & 16) ? 8 : 0;
    int prow0 = pbase + 16 * wave + rb;
    float rkv[8];
#pragma unroll
    for (int r = 0; r < 8; ++r) rkv[r] = rk[prow0 + r];
    upd_argmax(acc0, rkv, prow0, best[0], bidx[0]);
    upd_argmax(acc1, rkv, prow0, best[1], bidx[1]);
    upd_argmax(acc2, rkv, prow0, best[2], bidx[2]);
    upd_argmax(acc3, rkv, prow0, best[3], bidx[3]);
  }
  // merge lane pairs (n, n+16) sharing the same q column
#pragma unroll
  for (int t = 0; t < 4; ++t) {
    float ob = __shfl_xor(best[t], 16, 32);
    int   oi = __shfl_xor(bidx[t], 16, 32);
    if (ob > best[t]) { best[t] = ob; bidx[t] = oi; }
  }
  if (lane < 16) {
#pragma unroll
    for (int t = 0; t < 4; ++t) {
      redB[wave * QBLK + 16 * t + lane] = best[t];
      redI[wave * QBLK + 16 * t + lane] = bidx[t];
    }
  }
  __syncthreads();
  if (tid < QBLK) {
    float bb = redB[tid]; int bi = redI[tid];
#pragma unroll
    for (int w2 = 1; w2 < 8; ++w2) {
      float v = redB[w2 * QBLK + tid];
      if (v > bb) { bb = v; bi = redI[w2 * QBLK + tid]; }
    }
    size_t o = ((size_t)b * PSPLIT + blockIdx.y) * N_ + qbase + tid;
    bestOut[o] = bb; idxOut[o] = bi;
  }
}

__global__ void merge_argmax_kernel(const float* bestP, const int* idxP,
                                    int* idxF) {
  int i = blockIdx.x * 256 + threadIdx.x;
  if (i >= B_ * N_) return;
  int b = i / N_, q = i % N_;
  float bb = -1e30f; int bi = 0;
#pragma unroll
  for (int s = 0; s < PSPLIT; ++s) {
    size_t o = ((size_t)b * PSPLIT + s) * N_ + q;
    float v = bestP[o];
    if (v > bb) { bb = v; bi = idxP[o]; }
  }
  idxF[i] = bi;
}

// Overlap-add of best-matching whitened-style patches + count normalization.
__global__ void reassemble_kernel(const unsigned short* spad, const int* idxF,
                                  unsigned short* Rbf) {
  size_t i = (size_t)blockIdx.x * 256 + threadIdx.x;
  if (i >= (size_t)B_ * C_ * N_) return;
  int b = (int)(i / ((size_t)C_ * N_));
  int rem = (int)(i % ((size_t)C_ * N_));
  int c = rem / N_, q = rem % N_;
  int yo = q >> 6, xo = q & 63;
  const unsigned short* sp = spad + (size_t)b * C_ * PLANE_;
  const int* idx = idxF + (size_t)b * N_;
  float sum = 0.f, cnt = 0.f;
#pragma unroll
  for (int oy = 0; oy < 3; ++oy) {
    int y = yo + 1 - oy;
    if ((unsigned)y >= 64u) continue;
#pragma unroll
    for (int ox = 0; ox < 3; ++ox) {
      int x = xo + 1 - ox;
      if ((unsigned)x >= 64u) continue;
      int p = idx[y * 64 + x], py = p >> 6, px = p & 63;
      sum += bf2f(sp[((size_t)c * HP_ + py + oy) * HP_ + px + ox]);
      cnt += 1.f;
    }
  }
  Rbf[i] = f2bf(sum / cnt);
}

// out = sqrt(trS) * (Y_style @ R) + style mean   (cov^{1/2} recolor)
__global__ __launch_bounds__(256)
void color_apply_kernel(const float* Y, const float* tr,
                        const unsigned short* Rbf, const float* meanS,
                        float* out) {
  int b = blockIdx.z;
  int matS = b * 2 + 1;
  const float* Yb = Y + (size_t)matS * C_ * C_;
  const unsigned short* Rb = Rbf + (size_t)b * C_ * N_;
  float scale = sqrtf(tr[matS]);
  int wave = threadIdx.x >> 5;
  int tile = blockIdx.x * 8 + wave;
  int tm = tile >> 8, tn = tile & 255;
  int tn0 = (blockIdx.x * 8) & 255;

  __shared__ unsigned short ldsB[MSTEP * 128];

  v8f acc = {};
  for (int k0 = 0; k0 < C_; k0 += MSTEP) {
    if (threadIdx.x < 32) {
      tdm_load_2d_u16((unsigned)(size_t)&ldsB[0],
                      Rb + (size_t)k0 * N_ + (size_t)tn0 * 16,
                      /*rows=*/MSTEP, /*cols=*/128,
                      /*stride=*/N_, /*tensor_rows=*/C_);
      __builtin_amdgcn_s_wait_tensorcnt(0);
    }
    __syncthreads();
    v16bf a = frag_a_f32(Yb + (size_t)(tm * 16) * C_ + k0, C_);
    v16bf bfr = frag_b_nn_u16(ldsB + (tn - tn0) * 16, 128);
    acc = wmma_bf16(a, bfr, acc);
    __syncthreads();
  }
  int lane = threadIdx.x & 31;
  int col  = tn * 16 + (lane & 15);
  int rb   = (lane & 16) ? 8 : 0;
#pragma unroll
  for (int r = 0; r < 8; ++r) {
    int c = tm * 16 + rb + r;
    out[((size_t)b * C_ + c) * N_ + col] = acc[r] * scale + meanS[b * C_ + c];
  }
}

// ---------------------------------------------------------------------------
extern "C" void kernel_launch(void* const* d_in, const int* in_sizes, int n_in,
                              void* d_out, int out_size, void* d_ws, size_t ws_size,
                              hipStream_t stream) {
  (void)in_sizes; (void)n_in; (void)out_size; (void)ws_size;
  const float* content = (const float*)d_in[0];
  const float* style   = (const float*)d_in[1];

  size_t off = 0;
  auto alloc = [&](size_t bytes) -> void* {
    void* p = (char*)d_ws + off;
    off += (bytes + 255) & ~(size_t)255;
    return p;
  };
  unsigned short* fc_bf = (unsigned short*)alloc((size_t)B_ * C_ * N_ * 2);
  unsigned short* fs_bf = (unsigned short*)alloc((size_t)B_ * C_ * N_ * 2);
  float* meanC = (float*)alloc((size_t)B_ * C_ * 4);
  float* meanS = (float*)alloc((size_t)B_ * C_ * 4);
  float* cov   = (float*)alloc((size_t)4 * C_ * C_ * 4);
  float* tr    = (float*)alloc(4 * 4);
  float* Ya = (float*)alloc((size_t)4 * C_ * C_ * 4);
  float* Yb = (float*)alloc((size_t)4 * C_ * C_ * 4);
  float* Za = (float*)alloc((size_t)4 * C_ * C_ * 4);
  float* Zb = (float*)alloc((size_t)4 * C_ * C_ * 4);
  float* Tm = (float*)alloc((size_t)4 * C_ * C_ * 4);
  unsigned short* cpad = (unsigned short*)alloc((size_t)B_ * C_ * PLANE_ * 2);
  unsigned short* spad = (unsigned short*)alloc((size_t)B_ * C_ * PLANE_ * 2);
  float* rknorm = (float*)alloc((size_t)B_ * N_ * 4);
  float* bestP  = (float*)alloc((size_t)B_ * PSPLIT * N_ * 4);
  int*   idxP   = (int*)alloc((size_t)B_ * PSPLIT * N_ * 4);
  int*   idxF   = (int*)alloc((size_t)B_ * N_ * 4);
  unsigned short* Rbf = (unsigned short*)alloc((size_t)B_ * C_ * N_ * 2);

  size_t padElems = (size_t)B_ * C_ * PLANE_;
  fill_u16_kernel<<<(unsigned)((padElems + 255) / 256), 256, 0, stream>>>(cpad, padElems);
  fill_u16_kernel<<<(unsigned)((padElems + 255) / 256), 256, 0, stream>>>(spad, padElems);

  mean_center_kernel<<<dim3(B_ * C_, 2), 256, 0, stream>>>(
      content, style, fc_bf, fs_bf, meanC, meanS);

  cov_kernel<<<dim3(128, 1, 4), 256, 0, stream>>>(fc_bf, fs_bf, cov);
  trace_kernel<<<4, 256, 0, stream>>>(cov, tr);
  init_yz_kernel<<<(4 * C_ * C_ + 255) / 256, 256, 0, stream>>>(cov, tr, Ya, Za);

  float *Yc = Ya, *Zc = Za, *Yn = Yb, *Zn = Zb;
  for (int it = 0; it < NS_ITERS; ++it) {
    ns_matmul_kernel<<<dim3(128, 1, 4), 256, 0, stream>>>(Zc, Yc, Tm, -0.5f, 1.5f);
    ns_matmul_kernel<<<dim3(128, 1, 4), 256, 0, stream>>>(Yc, Tm, Yn, 1.0f, 0.0f);
    ns_matmul_kernel<<<dim3(128, 1, 4), 256, 0, stream>>>(Tm, Zc, Zn, 1.0f, 0.0f);
    float* t0 = Yc; Yc = Yn; Yn = t0;
    float* t1 = Zc; Zc = Zn; Zn = t1;
  }

  whiten_apply_kernel<<<dim3(1024, 1, 4), 256, 0, stream>>>(
      Zc, tr, fc_bf, fs_bf, cpad, spad);

  knorm_kernel<<<B_ * N_, 128, 0, stream>>>(spad, rknorm);

  score_argmax_kernel<<<dim3(N_ / QBLK, PSPLIT, B_), 256, 0, stream>>>(
      cpad, spad, rknorm, bestP, idxP);
  merge_argmax_kernel<<<(B_ * N_ + 255) / 256, 256, 0, stream>>>(bestP, idxP, idxF);

  reassemble_kernel<<<(unsigned)(((size_t)B_ * C_ * N_ + 255) / 256), 256, 0, stream>>>(
      spad, idxF, Rbf);

  color_apply_kernel<<<dim3(1024, 1, 2), 256, 0, stream>>>(
      Yc, tr, Rbf, meanS, (float*)d_out);
}
